// WindowAttentionHead_33621003993405
// MI455X (gfx1250) — matrix operand order repaired and checked by reading, not compiled, not run on hardware
//
#include <hip/hip_runtime.h>
#include <hip/hip_bf16.h>

typedef __attribute__((ext_vector_type(16))) _Float16 v16h;
typedef __attribute__((ext_vector_type(8)))  _Float16 v8h;
typedef __attribute__((ext_vector_type(8)))  float    v8f;

#define WA_N   49
#define WA_C   128
#define WA_H   4
#define WA_HD  32
#define WA_NP  64      // padded token count (4 x 16)
#define WA_XS  136     // row stride (halves) for x / q / k / out tiles (272B, 16B-aligned)
#define WA_VS  72      // vT row stride (halves)
#define WA_AS  72      // attn logits row stride (f32 elems); f16 probs overlaid in place

// ---- WMMA helpers ---------------------------------------------------------

__device__ __forceinline__ v8f wmma_f16(v16h a, v16h b, v8f c) {
    return __builtin_amdgcn_wmma_f32_16x16x32_f16(
        /*neg_a=*/false, a, /*neg_b=*/false, b,
        /*c_mod=*/(short)0, c, /*reuse_a=*/false, /*reuse_b=*/false);
}

// A-fragment: 16x32 f16, rows advance by `stride`, K contiguous within a row.
// ISA layout: lanes 0-15 = M rows, VGPR0-3 hold K = 8*hs..8*hs+7, VGPR4-7 hold K = 16+8*hs..
__device__ __forceinline__ v16h load_frag_row16(const _Float16* p0, int stride, int lane) {
    int m  = lane & 15;
    int hs = lane >> 4;
    const _Float16* p = p0 + m * stride + 8 * hs;
    v16h a;
    ((v8h*)&a)[0] = *(const v8h*)(p);        // K = 8*hs .. 8*hs+7
    ((v8h*)&a)[1] = *(const v8h*)(p + 16);   // K = 16+8*hs .. 16+8*hs+7
    return a;
}

// B-fragment: 32x16 f16 (KxN). Source memory: per output column, K contiguous
// (row stride `stride` between columns). Elems 0-7 -> K 8*ks.., elems 8-15 -> K 16+8*ks..
__device__ __forceinline__ v16h load_frag_col16(const _Float16* p0, int stride, int lane) {
    int n  = lane & 15;
    int ks = lane >> 4;
    const _Float16* p = p0 + n * stride + 8 * ks;
    v16h b;
    ((v8h*)&b)[0] = *(const v8h*)(p);        // K = 8*ks .. +7
    ((v8h*)&b)[1] = *(const v8h*)(p + 16);   // K = 16+8*ks .. +7
    return b;
}

// ---- Prep: transpose+convert weights to f16, pre-gather relative bias -----

__global__ void wa_prep_kernel(const float* __restrict__ qkv_w,     // [128][384]
                               const float* __restrict__ proj_w,    // [128][128]
                               const float* __restrict__ bias_table,// [169][4]
                               const int*   __restrict__ rel_index, // [49*49]
                               _Float16* __restrict__ qkvwT,        // [384][128]
                               _Float16* __restrict__ projwT,       // [128][128] (transposed)
                               float*    __restrict__ biasW)        // [4][49][49]
{
    const int QKV_E  = 3 * WA_C * WA_C;           // 49152
    const int PROJ_E = WA_C * WA_C;               // 16384
    const int BIAS_E = WA_H * WA_N * WA_N;        // 9604
    int i = blockIdx.x * 256 + threadIdx.x;
    if (i < QKV_E) {
        int nc = i / WA_C, kc = i % WA_C;
        qkvwT[i] = (_Float16)qkv_w[kc * (3 * WA_C) + nc];
    } else if (i < QKV_E + PROJ_E) {
        int j = i - QKV_E;
        int nc = j / WA_C, kc = j % WA_C;
        projwT[j] = (_Float16)proj_w[kc * WA_C + nc];
    } else if (i < QKV_E + PROJ_E + BIAS_E) {
        int j = i - QKV_E - PROJ_E;
        int h = j / (WA_N * WA_N);
        int rem = j % (WA_N * WA_N);
        biasW[j] = bias_table[rel_index[rem] * WA_H + h];
    }
}

// ---- Main fused window-attention kernel: 1 block = 1 window ---------------

__global__ __launch_bounds__(256)
void wa_attn_kernel(const float* __restrict__ x,       // [B][49][128]
                    const float* __restrict__ mask,    // [nW][49][49]
                    const float* __restrict__ qkv_b,   // [384]
                    const float* __restrict__ proj_b,  // [128]
                    const _Float16* __restrict__ qkvwT,  // [384][128]
                    const _Float16* __restrict__ projwT, // [128][128]
                    const float* __restrict__ biasW,     // [4][49][49]
                    float* __restrict__ out,             // [B][49][128]
                    int nW)
{
    __shared__ alignas(16) _Float16 sm_x [WA_NP * WA_XS];        // x (f16); reused as attn-out
    __shared__ alignas(16) _Float16 sm_q [WA_NP * WA_XS];        // q, pre-scaled
    __shared__ alignas(16) _Float16 sm_k [WA_NP * WA_XS];        // k, row-major
    __shared__ alignas(16) _Float16 sm_vT[WA_C  * WA_VS];        // v transposed: [chan][token]
    __shared__ alignas(16) float    sm_at[WA_H * WA_NP * WA_AS]; // logits; f16 probs overlaid

    const int tid  = threadIdx.x;
    const int lane = tid & 31;
    const int wv   = tid >> 5;
    const int b    = blockIdx.x;
    const float scale = 0.17677669529663687f;   // 32^-0.5

    // Load x -> f16 LDS, zero padding rows.
    for (int i = tid; i < WA_NP * WA_XS; i += 256) sm_x[i] = (_Float16)0.f;
    __syncthreads();
    const float* xg = x + (size_t)b * (WA_N * WA_C);
    for (int i = tid; i < WA_N * WA_C; i += 256) {
        int r = i >> 7, c = i & 127;
        sm_x[r * WA_XS + c] = (_Float16)xg[i];
    }
    __syncthreads();

    // ---- Stage A: QKV = x @ qkv_w + b   (M=64, N=384, K=128) -> 96 tiles
    for (int t = wv; t < 96; t += 8) {
        int mt = t & 3, nt = t >> 2;             // mt invariant per wave -> A frags hoisted
        int colg = nt * 16 + (lane & 15);
        float bb = qkv_b[colg];
        v8f acc;
        #pragma unroll
        for (int r = 0; r < 8; ++r) acc[r] = bb;
        #pragma unroll
        for (int kt = 0; kt < 4; ++kt) {
            v16h a  = load_frag_row16(sm_x + mt * 16 * WA_XS + kt * 32, WA_XS, lane);
            v16h bf = load_frag_col16(qkvwT + (size_t)(nt * 16) * WA_C + kt * 32, WA_C, lane);
            acc = wmma_f16(a, bf, acc);
        }
        int hs = lane >> 4;
        if (colg < WA_C) {                       // Q (scaled)
            #pragma unroll
            for (int r = 0; r < 8; ++r) {
                int row = mt * 16 + r + 8 * hs;
                sm_q[row * WA_XS + colg] = (_Float16)(acc[r] * scale);
            }
        } else if (colg < 2 * WA_C) {            // K
            #pragma unroll
            for (int r = 0; r < 8; ++r) {
                int row = mt * 16 + r + 8 * hs;
                sm_k[row * WA_XS + (colg - WA_C)] = (_Float16)acc[r];
            }
        } else {                                 // V (transposed -> contiguous b128 store)
            #pragma unroll
            for (int r = 0; r < 8; ++r) {
                int row = mt * 16 + r + 8 * hs;
                sm_vT[(colg - 2 * WA_C) * WA_VS + row] = (_Float16)acc[r];
            }
        }
    }
    __syncthreads();

    const float* maskp = mask + (size_t)(b % nW) * (WA_N * WA_N);

    // ---- Stage B: all heads, attn = q_h @ k_h^T + bias + mask -> 64 tiles
    for (int t = wv; t < 64; t += 8) {
        int h   = t >> 4;
        int rem = t & 15;
        int mt  = rem & 3, nt = rem >> 2;
        int mk = nt * 16 + (lane & 15);          // key index
        int hs = lane >> 4;
        v8f acc;
        #pragma unroll
        for (int r = 0; r < 8; ++r) {
            int nq = mt * 16 + r + 8 * hs;       // query index
            float v = 0.f;
            if (nq < WA_N && mk < WA_N) {
                int idx = nq * WA_N + mk;
                v = biasW[h * (WA_N * WA_N) + idx] + maskp[idx];
            }
            acc[r] = v;
        }
        v16h a  = load_frag_row16(sm_q + mt * 16 * WA_XS + h * WA_HD, WA_XS, lane);
        v16h bf = load_frag_col16(sm_k + nt * 16 * WA_XS + h * WA_HD, WA_XS, lane);
        acc = wmma_f16(a, bf, acc);
        float* arow = sm_at + (size_t)(h * WA_NP) * WA_AS + nt * 16 + (lane & 15);
        #pragma unroll
        for (int r = 0; r < 8; ++r) {
            int nq = mt * 16 + r + 8 * hs;
            arow[nq * WA_AS] = acc[r];
        }
    }
    __syncthreads();

    // ---- Stage C: softmax. 256 rows (4 heads x 64) -> exactly one per thread.
    // f16 probabilities are written in place over the f32 logits (byte 2j vs 4j:
    // ascending j only clobbers already-consumed elements within this thread's row).
    {
        int h = tid >> 6, row = tid & 63;
        float* ar = sm_at + (size_t)(h * WA_NP + row) * WA_AS;
        _Float16* ah = (_Float16*)ar;
        if (row < WA_N) {
            float mx = -1e30f;
            for (int j = 0; j < WA_N; ++j) mx = fmaxf(mx, ar[j]);
            float s = 0.f;
            for (int j = 0; j < WA_N; ++j) s += __expf(ar[j] - mx);
            float inv = 1.f / s;
            for (int j = 0; j < WA_N; ++j)
                ah[j] = (_Float16)(__expf(ar[j] - mx) * inv);
            for (int j = WA_N; j < WA_NP; ++j) ah[j] = (_Float16)0.f;
        } else {
            for (int j = 0; j < WA_NP; ++j) ah[j] = (_Float16)0.f;
        }
    }
    __syncthreads();

    // ---- Stage D: all heads, out_h = attn @ v_h -> 32 tiles (4 per wave)
    // Probs live at the front of each 288-byte logit row => row stride 144 halves.
    for (int t = wv; t < 32; t += 8) {
        int h   = t >> 3;
        int rem = t & 7;
        int mt  = rem & 3, ct = rem >> 2;
        const _Float16* ahb = (const _Float16*)(sm_at + (size_t)(h * WA_NP) * WA_AS);
        v8f acc;
        #pragma unroll
        for (int r = 0; r < 8; ++r) acc[r] = 0.f;
        #pragma unroll
        for (int kt = 0; kt < 2; ++kt) {
            v16h a  = load_frag_row16(ahb + mt * 16 * (2 * WA_AS) + kt * 32, 2 * WA_AS, lane);
            v16h bf = load_frag_col16(sm_vT + (size_t)(h * WA_HD + ct * 16) * WA_VS + kt * 32,
                                      WA_VS, lane);
            acc = wmma_f16(a, bf, acc);
        }
        int hs = lane >> 4;
        int colg = h * WA_HD + ct * 16 + (lane & 15);
        #pragma unroll
        for (int r = 0; r < 8; ++r) {
            int row = mt * 16 + r + 8 * hs;
            sm_x[row * WA_XS + colg] = (_Float16)acc[r];   // sm_x reused as attn-out
        }
    }
    __syncthreads();

    // ---- Stage E: result = attn_out @ proj_w + b   (M=64, N=128, K=128) -> 32 tiles
    float* og = out + (size_t)b * (WA_N * WA_C);
    for (int t = wv; t < 32; t += 8) {
        int mt = t & 3, nt = t >> 2;             // mt invariant per wave -> A frags hoisted
        int colg = nt * 16 + (lane & 15);
        float bb = proj_b[colg];
        v8f acc;
        #pragma unroll
        for (int r = 0; r < 8; ++r) acc[r] = bb;
        #pragma unroll
        for (int kt = 0; kt < 4; ++kt) {
            v16h a  = load_frag_row16(sm_x + mt * 16 * WA_XS + kt * 32, WA_XS, lane);
            v16h bf = load_frag_col16(projwT + (size_t)(nt * 16) * WA_C + kt * 32, WA_C, lane);
            acc = wmma_f16(a, bf, acc);
        }
        int hs = lane >> 4;
        #pragma unroll
        for (int r = 0; r < 8; ++r) {
            int row = mt * 16 + r + 8 * hs;
            if (row < WA_N) og[row * WA_C + colg] = acc[r];
        }
    }
}

// ---------------------------------------------------------------------------

extern "C" void kernel_launch(void* const* d_in, const int* in_sizes, int n_in,
                              void* d_out, int out_size, void* d_ws, size_t ws_size,
                              hipStream_t stream) {
    const float* x          = (const float*)d_in[0];
    const float* mask       = (const float*)d_in[1];
    const float* qkv_w      = (const float*)d_in[2];
    const float* qkv_b      = (const float*)d_in[3];
    const float* proj_w     = (const float*)d_in[4];
    const float* proj_b     = (const float*)d_in[5];
    const float* bias_table = (const float*)d_in[6];
    const int*   rel_index  = (const int*)d_in[7];

    const int Bw = in_sizes[0] / (WA_N * WA_C);   // 4096 windows
    const int nW = in_sizes[1] / (WA_N * WA_N);   // 64

    // Workspace layout (16B aligned offsets)
    char* ws = (char*)d_ws;
    _Float16* qkvwT  = (_Float16*)(ws);            // 3*128*128 f16 = 98304 B
    _Float16* projwT = (_Float16*)(ws + 98304);    // 128*128 f16   = 32768 B
    float*    biasW  = (float*)   (ws + 131072);   // 4*49*49 f32   = 38416 B

    const int prep_total = 3 * WA_C * WA_C + WA_C * WA_C + WA_H * WA_N * WA_N; // 75140
    wa_prep_kernel<<<(prep_total + 255) / 256, 256, 0, stream>>>(
        qkv_w, proj_w, bias_table, rel_index, qkvwT, projwT, biasW);

    wa_attn_kernel<<<Bw, 256, 0, stream>>>(
        x, mask, qkv_b, proj_b, qkvwT, projwT, biasW, (float*)d_out, nW);
}
